// SAINT_45165876085306
// MI455X (gfx1250) — compile-verified, matrix operand order checked by reading, not run on hardware
//
#include <hip/hip_runtime.h>

#define LSEQ   1024
#define DMODEL 256
#define FFDIM  2048
#define NSKILLS 1000
#define KPAD   1024
#define NLAYERS 6

typedef __bf16 bf16;
typedef __attribute__((ext_vector_type(16))) __bf16 v16bf;
typedef __attribute__((ext_vector_type(8)))  __bf16 v8bf;
typedef __attribute__((ext_vector_type(8)))  float  v8f;

// ---------------------------------------------------------------------------
// bf16 WMMA GEMM:  C = alpha * A@B^T(+bias)(+resid), optional ReLU.
//   A : [M,K] row-major bf16 (lda = row stride)
//   Bt: [N,K] row-major bf16 (ldb = row stride)  == B^T, so C = A x Bt^T
// Outputs: Cf (f32, [M,N]) and/or Cb (bf16), either may be null.
// transCb: bf16 output written transposed as [N,M] (used to produce V^T).
//
// One wave computes a 32x32 tile: 2 A-frags x 2 B-frags -> 4 WMMAs / K-step.
// Per CDNA5 ISA 7.12.2 (wave32), a 16x32 bf16 A fragment for lane
// (row = lane&15, half = lane>>4) is two contiguous 8xbf16 runs at
// k0 + half*8 and k0 + 16 + half*8  ->  two global_load_b128 per fragment.
// B fragment is identical with lane indexing the output column (row of Bt).
// ---------------------------------------------------------------------------
__device__ __forceinline__ v16bf load_frag(const bf16* __restrict__ p) {
  v8bf lo = *(const v8bf*)(p);
  v8bf hi = *(const v8bf*)(p + 16);
  return __builtin_shufflevector(lo, hi, 0, 1, 2, 3, 4, 5, 6, 7,
                                 8, 9, 10, 11, 12, 13, 14, 15);
}

__global__ __launch_bounds__(256) void gemm_wmma_bf16(
    const bf16* __restrict__ A, int lda,
    const bf16* __restrict__ Bt, int ldb,
    const float* __restrict__ bias,
    const float* __restrict__ resid,
    float* __restrict__ Cf, bf16* __restrict__ Cb,
    int M, int N, int K, float alpha, int relu, int transCb)
{
  const int lane    = threadIdx.x & 31;
  const int wave    = threadIdx.x >> 5;
  const int tiles_n = N >> 5;
  const int w       = blockIdx.x * 8 + wave;
  const int tm      = (w / tiles_n) << 5;
  const int tn      = (w % tiles_n) << 5;
  const int row     = lane & 15;   // A row / B column within 16-tile
  const int half    = lane >> 4;

  v8f acc00 = {}, acc01 = {}, acc10 = {}, acc11 = {};

  const bf16* __restrict__ Ar0 = A  + (size_t)(tm + row) * lda;
  const bf16* __restrict__ Ar1 = Ar0 + (size_t)16 * lda;
  const bf16* __restrict__ Br0 = Bt + (size_t)(tn + row) * ldb;
  const bf16* __restrict__ Br1 = Br0 + (size_t)16 * ldb;

  for (int k0 = 0; k0 < K; k0 += 32) {
    const int o = k0 + half * 8;
    const v16bf a0 = load_frag(Ar0 + o);
    const v16bf a1 = load_frag(Ar1 + o);
    const v16bf b0 = load_frag(Br0 + o);
    const v16bf b1 = load_frag(Br1 + o);
    acc00 = __builtin_amdgcn_wmma_f32_16x16x32_bf16(false, a0, false, b0,
                                                    (short)0, acc00, false, false);
    acc01 = __builtin_amdgcn_wmma_f32_16x16x32_bf16(false, a0, false, b1,
                                                    (short)0, acc01, false, false);
    acc10 = __builtin_amdgcn_wmma_f32_16x16x32_bf16(false, a1, false, b0,
                                                    (short)0, acc10, false, false);
    acc11 = __builtin_amdgcn_wmma_f32_16x16x32_bf16(false, a1, false, b1,
                                                    (short)0, acc11, false, false);
  }

  v8f* accs[4] = { &acc00, &acc01, &acc10, &acc11 };
#pragma unroll
  for (int mi = 0; mi < 2; ++mi) {
#pragma unroll
    for (int ni = 0; ni < 2; ++ni) {
      const v8f& acc = *accs[mi * 2 + ni];
      const int col  = tn + ni * 16 + row;
      const float bv = bias ? bias[col] : 0.0f;
#pragma unroll
      for (int r = 0; r < 8; ++r) {
        const int m = tm + mi * 16 + r + 8 * half;
        const size_t idx = (size_t)m * N + col;
        float v = alpha * acc[r] + bv;
        if (resid) v += resid[idx];
        if (relu)  v = v > 0.0f ? v : 0.0f;
        if (Cf) Cf[idx] = v;
        if (Cb) Cb[transCb ? ((size_t)col * M + m) : idx] = (bf16)v;
      }
    }
  }
}

// ---------------------------------------------------------------------------
// Causal softmax over rows of S (f32) -> probabilities as bf16.
// ---------------------------------------------------------------------------
__global__ __launch_bounds__(256) void softmax_causal_k(
    const float* __restrict__ S, bf16* __restrict__ P)
{
  __shared__ float red[256];
  const int r = blockIdx.x, t = threadIdx.x;
  float vbuf[4], e[4];
  float mx = -1e30f;
#pragma unroll
  for (int i = 0; i < 4; ++i) {
    const int c = t + i * 256;
    const float v = (c <= r) ? S[(size_t)r * LSEQ + c] : -1e30f;
    vbuf[i] = v;
    mx = v > mx ? v : mx;
  }
  red[t] = mx; __syncthreads();
  for (int s = 128; s > 0; s >>= 1) {
    if (t < s && red[t + s] > red[t]) red[t] = red[t + s];
    __syncthreads();
  }
  mx = red[0]; __syncthreads();
  float sum = 0.0f;
#pragma unroll
  for (int i = 0; i < 4; ++i) {
    const int c = t + i * 256;
    e[i] = (c <= r) ? __expf(vbuf[i] - mx) : 0.0f;
    sum += e[i];
  }
  red[t] = sum; __syncthreads();
  for (int s = 128; s > 0; s >>= 1) {
    if (t < s) red[t] += red[t + s];
    __syncthreads();
  }
  const float inv = 1.0f / red[0];
#pragma unroll
  for (int i = 0; i < 4; ++i) {
    const int c = t + i * 256;
    P[(size_t)r * LSEQ + c] = (bf16)(e[i] * inv);
  }
}

// ---------------------------------------------------------------------------
// Row LayerNorm over D=256 (blockDim == D). Writes f32 and/or bf16.
// ---------------------------------------------------------------------------
__global__ __launch_bounds__(256) void layernorm_k(
    const float* __restrict__ X, const float* __restrict__ g,
    const float* __restrict__ b, float* __restrict__ Yf, bf16* __restrict__ Yb)
{
  __shared__ float red[256];
  const int r = blockIdx.x, t = threadIdx.x;
  const float x = X[(size_t)r * DMODEL + t];
  red[t] = x; __syncthreads();
  for (int s = 128; s > 0; s >>= 1) { if (t < s) red[t] += red[t + s]; __syncthreads(); }
  const float mean = red[0] * (1.0f / DMODEL); __syncthreads();
  const float d = x - mean;
  red[t] = d * d; __syncthreads();
  for (int s = 128; s > 0; s >>= 1) { if (t < s) red[t] += red[t + s]; __syncthreads(); }
  const float var = red[0] * (1.0f / DMODEL);
  const float y = d * rsqrtf(var + 1e-5f) * g[t] + b[t];
  if (Yf) Yf[(size_t)r * DMODEL + t] = y;
  if (Yb) Yb[(size_t)r * DMODEL + t] = (bf16)y;
}

// ---------------------------------------------------------------------------
// Transposing f32 -> bf16 weight conversion: src [K,N] f32 -> dst [N,K] bf16.
// 32x32 tile via LDS, block (32,8). K, N multiples of 32.
// ---------------------------------------------------------------------------
__global__ __launch_bounds__(256) void convT_k(
    const float* __restrict__ src, bf16* __restrict__ dst, int K, int N)
{
  __shared__ float tile[32][33];
  const int bk = blockIdx.x * 32, bn = blockIdx.y * 32;
  const int tx = threadIdx.x, ty = threadIdx.y;
  for (int i = ty; i < 32; i += 8)
    tile[i][tx] = src[(size_t)(bk + i) * N + bn + tx];
  __syncthreads();
  for (int i = ty; i < 32; i += 8)
    dst[(size_t)(bn + i) * K + bk + tx] = (bf16)tile[tx][i];
}

// Multi-hot A matrix: A[l][s] = Q[items[l], s] (zero-padded to KPAD cols)
__global__ __launch_bounds__(256) void build_itemA_k(
    const int* __restrict__ items, const int* __restrict__ Q, bf16* __restrict__ A)
{
  const int l = blockIdx.x;
  const int it = items[l];
  for (int s = threadIdx.x; s < KPAD; s += 256) {
    float v = 0.0f;
    if (s < NSKILLS) v = (float)Q[(size_t)it * NSKILLS + s];
    A[(size_t)l * KPAD + s] = (bf16)v;
  }
}

// skill_embedding[user]^T -> bf16 [DMODEL, KPAD] (zero-padded cols >= NSKILLS)
__global__ __launch_bounds__(256) void build_skillBT_k(
    const int* __restrict__ user, const float* __restrict__ emb, bf16* __restrict__ Bt)
{
  const int d0 = blockIdx.x;          // 0..DMODEL-1
  const int u = user[0];
  for (int s = threadIdx.x; s < KPAD; s += 256) {
    float v = 0.0f;
    if (s < NSKILLS)
      v = emb[(size_t)u * NSKILLS * DMODEL + (size_t)s * DMODEL + d0];
    Bt[(size_t)d0 * KPAD + s] = (bf16)v;
  }
}

// tgt = concat(start_token[user], answer_embedding[user][answers[:-1]])
__global__ __launch_bounds__(256) void build_tgt_k(
    const int* __restrict__ user, const int* __restrict__ answers,
    const float* __restrict__ ansemb, const float* __restrict__ start,
    float* __restrict__ Yf, bf16* __restrict__ Yb)
{
  const int l = blockIdx.x, d0 = threadIdx.x;
  const int u = user[0];
  float v;
  if (l == 0) v = start[(size_t)u * DMODEL + d0];
  else        v = ansemb[(size_t)u * 2 * DMODEL + (size_t)answers[l - 1] * DMODEL + d0];
  Yf[(size_t)l * DMODEL + d0] = v;
  Yb[(size_t)l * DMODEL + d0] = (bf16)v;
}

// out[r] = sigmoid(Y[r,:] @ w + b)
__global__ __launch_bounds__(256) void pred_sigmoid_k(
    const float* __restrict__ Y, const float* __restrict__ w,
    const float* __restrict__ b, float* __restrict__ out)
{
  __shared__ float red[256];
  const int r = blockIdx.x, t = threadIdx.x;
  red[t] = Y[(size_t)r * DMODEL + t] * w[t]; __syncthreads();
  for (int s = 128; s > 0; s >>= 1) { if (t < s) red[t] += red[t + s]; __syncthreads(); }
  if (t == 0) out[r] = 1.0f / (1.0f + __expf(-(red[0] + b[0])));
}

// ---------------------------------------------------------------------------
// Host orchestration
// ---------------------------------------------------------------------------
extern "C" void kernel_launch(void* const* d_in, const int* in_sizes, int n_in,
                              void* d_out, int out_size, void* d_ws, size_t ws_size,
                              hipStream_t stream)
{
  (void)in_sizes; (void)n_in; (void)out_size; (void)ws_size;

  // --- flattened input indices (setup_inputs dict order, leaf order) ---
  const int IN_USER = 0, IN_ITEMS = 1, IN_ANS = 2, IN_Q = 3, IN_SKEMB = 4,
            IN_AEMB = 5, IN_START = 6;
  const int ENC0 = 7;                       // 16 arrays per encoder layer
  const int DEC0 = ENC0 + 16 * NLAYERS;     // 26 arrays per decoder layer
  const int POST = DEC0 + 26 * NLAYERS;
  const int IN_ENC_G = POST + 0, IN_ENC_B = POST + 1;
  const int IN_DEC_G = POST + 2, IN_DEC_B = POST + 3;
  const int IN_PW = POST + 4, IN_PB = POST + 5;

  const int* d_user  = (const int*)d_in[IN_USER];
  const int* d_items = (const int*)d_in[IN_ITEMS];
  const int* d_ans   = (const int*)d_in[IN_ANS];
  const int* d_Q     = (const int*)d_in[IN_Q];
  const float* d_skemb = (const float*)d_in[IN_SKEMB];
  const float* d_aemb  = (const float*)d_in[IN_AEMB];
  const float* d_start = (const float*)d_in[IN_START];

  // --- workspace bump allocator ---
  size_t off = 0;
  auto walloc = [&](size_t bytes) -> void* {
    off = (off + 255) & ~(size_t)255;
    void* p = (char*)d_ws + off;
    off += bytes;
    return p;
  };

  // bf16 transposed weights ([N,K] layout)
  bf16 *eWq[NLAYERS], *eWk[NLAYERS], *eWv[NLAYERS], *eWo[NLAYERS],
       *eW1[NLAYERS], *eW2[NLAYERS];
  bf16 *dsWq[NLAYERS], *dsWk[NLAYERS], *dsWv[NLAYERS], *dsWo[NLAYERS],
       *dcWq[NLAYERS], *dcWk[NLAYERS], *dcWv[NLAYERS], *dcWo[NLAYERS],
       *dW1[NLAYERS], *dW2[NLAYERS];
  const size_t DD = (size_t)DMODEL * DMODEL, DF = (size_t)DMODEL * FFDIM;
  for (int i = 0; i < NLAYERS; ++i) {
    eWq[i] = (bf16*)walloc(DD * 2); eWk[i] = (bf16*)walloc(DD * 2);
    eWv[i] = (bf16*)walloc(DD * 2); eWo[i] = (bf16*)walloc(DD * 2);
    eW1[i] = (bf16*)walloc(DF * 2); eW2[i] = (bf16*)walloc(DF * 2);
  }
  for (int i = 0; i < NLAYERS; ++i) {
    dsWq[i] = (bf16*)walloc(DD * 2); dsWk[i] = (bf16*)walloc(DD * 2);
    dsWv[i] = (bf16*)walloc(DD * 2); dsWo[i] = (bf16*)walloc(DD * 2);
    dcWq[i] = (bf16*)walloc(DD * 2); dcWk[i] = (bf16*)walloc(DD * 2);
    dcWv[i] = (bf16*)walloc(DD * 2); dcWo[i] = (bf16*)walloc(DD * 2);
    dW1[i]  = (bf16*)walloc(DF * 2); dW2[i]  = (bf16*)walloc(DF * 2);
  }

  // activations
  bf16* itemA   = (bf16*)walloc((size_t)LSEQ * KPAD * 2);
  bf16* skillBt = (bf16*)walloc((size_t)DMODEL * KPAD * 2);
  float* xf   = (float*)walloc((size_t)LSEQ * DMODEL * 4);
  bf16*  xb   = (bf16*)walloc((size_t)LSEQ * DMODEL * 2);
  float* tf   = (float*)walloc((size_t)LSEQ * DMODEL * 4);
  float* yf   = (float*)walloc((size_t)LSEQ * DMODEL * 4);
  bf16*  yb   = (bf16*)walloc((size_t)LSEQ * DMODEL * 2);
  float* memf = (float*)walloc((size_t)LSEQ * DMODEL * 4);
  bf16*  memb = (bf16*)walloc((size_t)LSEQ * DMODEL * 2);
  bf16*  Qb   = (bf16*)walloc((size_t)LSEQ * DMODEL * 2);
  bf16*  Kb   = (bf16*)walloc((size_t)LSEQ * DMODEL * 2);
  bf16*  Vt   = (bf16*)walloc((size_t)DMODEL * LSEQ * 2);   // V^T [D, L]
  bf16*  Ob   = (bf16*)walloc((size_t)LSEQ * DMODEL * 2);
  float* Sf   = (float*)walloc((size_t)LSEQ * LSEQ * 4);
  bf16*  Pb   = (bf16*)walloc((size_t)LSEQ * LSEQ * 2);
  bf16*  hb   = (bf16*)walloc((size_t)LSEQ * FFDIM * 2);

  auto convT = [&](int idx, bf16* dst, int K, int N) {
    dim3 grid(K / 32, N / 32), blk(32, 8);
    convT_k<<<grid, blk, 0, stream>>>((const float*)d_in[idx], dst, K, N);
  };
  auto F = [&](int idx) { return (const float*)d_in[idx]; };

  // convert+transpose all weight matrices to bf16 [N,K]
  for (int i = 0; i < NLAYERS; ++i) {
    const int e = ENC0 + 16 * i;
    convT(e + 0, eWq[i], DMODEL, DMODEL); convT(e + 2, eWk[i], DMODEL, DMODEL);
    convT(e + 4, eWv[i], DMODEL, DMODEL); convT(e + 6, eWo[i], DMODEL, DMODEL);
    convT(e + 8, eW1[i], DMODEL, FFDIM);  convT(e + 10, eW2[i], FFDIM, DMODEL);
    const int d = DEC0 + 26 * i;
    convT(d + 0, dsWq[i], DMODEL, DMODEL); convT(d + 2, dsWk[i], DMODEL, DMODEL);
    convT(d + 4, dsWv[i], DMODEL, DMODEL); convT(d + 6, dsWo[i], DMODEL, DMODEL);
    convT(d + 8, dcWq[i], DMODEL, DMODEL); convT(d + 10, dcWk[i], DMODEL, DMODEL);
    convT(d + 12, dcWv[i], DMODEL, DMODEL); convT(d + 14, dcWo[i], DMODEL, DMODEL);
    convT(d + 16, dW1[i], DMODEL, FFDIM);   convT(d + 18, dW2[i], FFDIM, DMODEL);
  }

  // C = alpha * A[M,K] x Bt[N,K]^T
  auto gemm = [&](const bf16* A, int lda, const bf16* Bt, int ldb,
                  const float* bias, const float* resid,
                  float* Cf, bf16* Cb, int M, int N, int K,
                  float alpha, int relu, int transCb) {
    const int blocks = (M / 32) * (N / 32) / 8;
    gemm_wmma_bf16<<<blocks, 256, 0, stream>>>(A, lda, Bt, ldb, bias, resid,
                                               Cf, Cb, M, N, K, alpha, relu, transCb);
  };

  // attention: tOut = resid + (softmax_causal(QK^T/sqrt(D)) V) Wo + bo
  auto attn = [&](const bf16* qin, const bf16* kvin,
                  const bf16* Wq, const float* bq, const bf16* Wk, const float* bk,
                  const bf16* Wv, const float* bv, const bf16* Wo, const float* bo,
                  const float* residF, float* tOut) {
    gemm(qin,  DMODEL, Wq, DMODEL, bq, nullptr, nullptr, Qb,
         LSEQ, DMODEL, DMODEL, 1.f, 0, 0);
    gemm(kvin, DMODEL, Wk, DMODEL, bk, nullptr, nullptr, Kb,
         LSEQ, DMODEL, DMODEL, 1.f, 0, 0);
    gemm(kvin, DMODEL, Wv, DMODEL, bv, nullptr, nullptr, Vt,
         LSEQ, DMODEL, DMODEL, 1.f, 0, 1 /*write V^T*/);
    gemm(Qb, DMODEL, Kb, DMODEL, nullptr, nullptr, Sf, nullptr,
         LSEQ, LSEQ, DMODEL, 0.0625f /*1/sqrt(256)*/, 0, 0);
    softmax_causal_k<<<LSEQ, 256, 0, stream>>>(Sf, Pb);
    gemm(Pb, LSEQ, Vt, LSEQ, nullptr, nullptr, nullptr, Ob,
         LSEQ, DMODEL, LSEQ, 1.f, 0, 0);
    gemm(Ob, DMODEL, Wo, DMODEL, bo, residF, tOut, nullptr,
         LSEQ, DMODEL, DMODEL, 1.f, 0, 0);
  };

  // --- item embedding: x = (Q[items]) @ skill_emb[user] ---
  build_itemA_k<<<LSEQ, 256, 0, stream>>>(d_items, d_Q, itemA);
  build_skillBT_k<<<DMODEL, 256, 0, stream>>>(d_user, d_skemb, skillBt);
  gemm(itemA, KPAD, skillBt, KPAD, nullptr, nullptr, xf, xb,
       LSEQ, DMODEL, KPAD, 1.f, 0, 0);

  // --- encoder ---
  for (int i = 0; i < NLAYERS; ++i) {
    const int e = ENC0 + 16 * i;
    attn(xb, xb, eWq[i], F(e + 1), eWk[i], F(e + 3), eWv[i], F(e + 5),
         eWo[i], F(e + 7), xf, tf);
    layernorm_k<<<LSEQ, 256, 0, stream>>>(tf, F(e + 12), F(e + 13), xf, xb);
    gemm(xb, DMODEL, eW1[i], DMODEL, F(e + 9), nullptr, nullptr, hb,
         LSEQ, FFDIM, DMODEL, 1.f, 1, 0);
    gemm(hb, FFDIM, eW2[i], FFDIM, F(e + 11), xf, tf, nullptr,
         LSEQ, DMODEL, FFDIM, 1.f, 0, 0);
    layernorm_k<<<LSEQ, 256, 0, stream>>>(tf, F(e + 14), F(e + 15), xf, xb);
  }
  layernorm_k<<<LSEQ, 256, 0, stream>>>(xf, F(IN_ENC_G), F(IN_ENC_B), memf, memb);

  // --- decoder ---
  build_tgt_k<<<LSEQ, 256, 0, stream>>>(d_user, d_ans, d_aemb, d_start, yf, yb);
  for (int i = 0; i < NLAYERS; ++i) {
    const int d = DEC0 + 26 * i;
    // self-attention (causal)
    attn(yb, yb, dsWq[i], F(d + 1), dsWk[i], F(d + 3), dsWv[i], F(d + 5),
         dsWo[i], F(d + 7), yf, tf);
    layernorm_k<<<LSEQ, 256, 0, stream>>>(tf, F(d + 20), F(d + 21), yf, yb);
    // cross-attention (reference applies the same causal mask to memory)
    attn(yb, memb, dcWq[i], F(d + 9), dcWk[i], F(d + 11), dcWv[i], F(d + 13),
         dcWo[i], F(d + 15), yf, tf);
    layernorm_k<<<LSEQ, 256, 0, stream>>>(tf, F(d + 22), F(d + 23), yf, yb);
    // FFN
    gemm(yb, DMODEL, dW1[i], DMODEL, F(d + 17), nullptr, nullptr, hb,
         LSEQ, FFDIM, DMODEL, 1.f, 1, 0);
    gemm(hb, FFDIM, dW2[i], FFDIM, F(d + 19), yf, tf, nullptr,
         LSEQ, DMODEL, FFDIM, 1.f, 0, 0);
    layernorm_k<<<LSEQ, 256, 0, stream>>>(tf, F(d + 24), F(d + 25), yf, yb);
  }
  layernorm_k<<<LSEQ, 256, 0, stream>>>(yf, F(IN_DEC_G), F(IN_DEC_B), tf, nullptr);

  // --- prediction head ---
  pred_sigmoid_k<<<LSEQ, 256, 0, stream>>>(tf, F(IN_PW), F(IN_PB), (float*)d_out);
}
